// ParallelTransformerBlock_18966575579502
// MI455X (gfx1250) — matrix operand order repaired and checked by reading, not compiled, not run on hardware
//
#include <hip/hip_runtime.h>

// ---------------- problem constants ----------------
#define BATCH   2
#define NSEQ    2048
#define DIM     2048
#define HEADS   32
#define DHEAD   64
#define FFI     8192            // FF_INNER
#define FUSED   18560           // H*Dh + Dh + Dh + 2*FFI
#define MTOT    (BATCH * NSEQ)  // 4096 rows

// ---------------- types ----------------
typedef __attribute__((ext_vector_type(16))) __bf16        v16bf;
typedef __attribute__((ext_vector_type(8)))  float         v8f;
typedef __attribute__((ext_vector_type(4)))  unsigned int  v4u;
typedef __attribute__((ext_vector_type(4)))  unsigned int  su4;
typedef __attribute__((ext_vector_type(8)))  unsigned int  su8;

union Frag { v16bf v; v4u q[2]; };

// ---------------- helpers ----------------
__device__ __forceinline__ unsigned short f2b(float f) {
    unsigned int u = __float_as_uint(f);
    u += 0x7fffu + ((u >> 16) & 1u);     // round-to-nearest-even
    return (unsigned short)(u >> 16);
}
__device__ __forceinline__ float b2f(unsigned short h) {
    return __uint_as_float(((unsigned int)h) << 16);
}
__device__ __forceinline__ v8f zero8() {
    v8f z;
#pragma unroll
    for (int i = 0; i < 8; ++i) z[i] = 0.0f;
    return z;
}
__device__ __forceinline__ float rowmax16(float v) {
    v = fmaxf(v, __shfl_xor(v, 1, 32));
    v = fmaxf(v, __shfl_xor(v, 2, 32));
    v = fmaxf(v, __shfl_xor(v, 4, 32));
    v = fmaxf(v, __shfl_xor(v, 8, 32));
    return v;
}
__device__ __forceinline__ float rowsum16(float v) {
    v += __shfl_xor(v, 1, 32);
    v += __shfl_xor(v, 2, 32);
    v += __shfl_xor(v, 4, 32);
    v += __shfl_xor(v, 8, 32);
    return v;
}

// ---------------------------------------------------------------------
// Tensor Data Mover: load a 16-row x 32-halves tile (row stride K halves)
// into LDS at byte offset lds_off, padded 64B->80B rows (pitch 40 halves).
// D# layout per CDNA5 ISA ch.8 (group0: 4 SGPRs, group1: 8 SGPRs).
// ---------------------------------------------------------------------
__device__ __forceinline__ void tdm_load_tile16x32(const unsigned short* gaddr,
                                                   unsigned lds_off, int K) {
    unsigned long long ga = (unsigned long long)(uintptr_t)gaddr;
    su4 g0;
    g0[0] = 1u;                                                  // count=1, user mode
    g0[1] = lds_off;                                             // LDS byte address
    g0[2] = (unsigned)ga;                                        // global_addr[31:0]
    g0[3] = (unsigned)((ga >> 32) & 0x01FFFFFFu) | 0x80000000u;  // addr[56:32] | type=2
    su8 g1;
    g1[0] = (1u << 16)        // data_size = 1 -> 2 bytes
          | (1u << 20)        // pad_enable
          | (3u << 22)        // pad_interval: every 16 DWORDs (64B)
          | (3u << 25);       // pad_amount: 4 DWORDs (16B)  -> 80B pitch
    g1[1] = ((unsigned)K & 0xFFFFu) << 16;            // tensor_dim0[15:0]
    g1[2] = ((unsigned)K >> 16) | (16u << 16);        // tensor_dim0[31:16] | tensor_dim1=16
    g1[3] = (32u << 16);                              // tile_dim0 = 32 elements
    g1[4] = 16u;                                      // tile_dim1 = 16 rows, tile_dim2=0
    g1[5] = (unsigned)K;                              // tensor_dim0_stride[31:0]
    g1[6] = 0u;                                       // stride hi, dim1_stride lo
    g1[7] = 0u;
    asm volatile("tensor_load_to_lds %0, %1" :: "s"(g0), "s"(g1) : "memory");
}

// =====================================================================
// 1) LayerNorm (row of 2048) -> bf16
// =====================================================================
__global__ void ln_kernel(const float* __restrict__ x, const float* __restrict__ gamma,
                          unsigned short* __restrict__ xn) {
    const int row = blockIdx.x;
    const int tid = threadIdx.x;
    const float* xr = x + (size_t)row * DIM;
    float s = 0.f, s2 = 0.f;
    for (int i = tid; i < DIM; i += 256) { float v = xr[i]; s += v; s2 += v * v; }
    __shared__ float r1[256], r2[256];
    r1[tid] = s; r2[tid] = s2; __syncthreads();
    for (int off = 128; off > 0; off >>= 1) {
        if (tid < off) { r1[tid] += r1[tid + off]; r2[tid] += r2[tid + off]; }
        __syncthreads();
    }
    const float mu  = r1[0] * (1.0f / DIM);
    const float var = r2[0] * (1.0f / DIM) - mu * mu;
    const float rstd = rsqrtf(var + 1e-5f);
    for (int i = tid; i < DIM; i += 256)
        xn[(size_t)row * DIM + i] = f2b((xr[i] - mu) * rstd * gamma[i]);
}

// =====================================================================
// 2) f32 -> bf16 elementwise (weights)
// =====================================================================
__global__ void cvt_kernel(const float* __restrict__ in, unsigned short* __restrict__ out,
                           long long n) {
    long long base = (long long)blockIdx.x * 1024 + threadIdx.x;
#pragma unroll
    for (int t = 0; t < 4; ++t) {
        long long i = base + (long long)t * 256;
        if (i < n) out[i] = f2b(in[i]);
    }
}

// =====================================================================
// 3) GEMM: C[M,N] = A[M,K] * W[N,K]^T, bf16 in, WMMA f32 accumulate.
//    128x128 workgroup tile, 8 waves (4x2), wave tile 32x64, K-step 32.
//    Tiles staged to LDS by the Tensor Data Mover, double buffered.
// =====================================================================
__global__ void gemm_kernel(const unsigned short* __restrict__ A,
                            const unsigned short* __restrict__ W,
                            int K, float* __restrict__ Cf,
                            unsigned short* __restrict__ Cb, int ldc, int beta) {
    const int nBase = blockIdx.x * 128;
    const int mBase = blockIdx.y * 128;
    const int tid  = threadIdx.x;
    const int lane = tid & 31;
    const int wm = (tid >> 5) >> 1;   // 0..3 -> 32-row slab
    const int wn = (tid >> 5) & 1;    // 0..1 -> 64-col slab
    const int lr = lane & 15;
    const int lh = lane >> 4;

    __shared__ unsigned short lsA[2][128 * 40];   // 32 halves + 8 pad per row
    __shared__ unsigned short lsW[2][128 * 40];

    // scalar (per-wave) TDM setup: each wave DMAs its 16-row slice of A and W
    const int waveu = __builtin_amdgcn_readfirstlane(tid >> 5);
    const unsigned short* gA = A + (size_t)(mBase + waveu * 16) * K;
    const unsigned short* gW = W + (size_t)(nBase + waveu * 16) * K;
    const unsigned ldsA0 = (unsigned)(uintptr_t)&lsA[0][waveu * 16 * 40];
    const unsigned ldsA1 = (unsigned)(uintptr_t)&lsA[1][waveu * 16 * 40];
    const unsigned ldsW0 = (unsigned)(uintptr_t)&lsW[0][waveu * 16 * 40];
    const unsigned ldsW1 = (unsigned)(uintptr_t)&lsW[1][waveu * 16 * 40];

    v8f acc[2][4];
#pragma unroll
    for (int mt = 0; mt < 2; ++mt)
#pragma unroll
        for (int nt = 0; nt < 4; ++nt) acc[mt][nt] = zero8();

    // prologue: DMA first K-tile into buffer 0
    tdm_load_tile16x32(gA, ldsA0, K);
    tdm_load_tile16x32(gW, ldsW0, K);

    for (int kk = 0; kk < K; kk += 32) {
        const int buf = (kk >> 5) & 1;
        __builtin_amdgcn_s_wait_tensorcnt(0);  // my slices landed
        __syncthreads();                        // everyone's slices landed
        if (kk + 32 < K) {                      // async DMA of next K-tile
            tdm_load_tile16x32(gA + kk + 32, buf ? ldsA0 : ldsA1, K);
            tdm_load_tile16x32(gW + kk + 32, buf ? ldsW0 : ldsW1, K);
        }

        Frag fa[2], fb[4];
#pragma unroll
        for (int mt = 0; mt < 2; ++mt) {        // A: lane row lr, contiguous 16-half K run
            const unsigned short* base = &lsA[buf][(wm * 32 + mt * 16 + lr) * 40 + lh * 16];
            fa[mt].q[0] = *(const v4u*)base;
            fa[mt].q[1] = *(const v4u*)(base + 8);
        }
#pragma unroll
        for (int nt = 0; nt < 4; ++nt) {        // B: lane col lr, contiguous 16-half K run
            const unsigned short* base = &lsW[buf][(wn * 64 + nt * 16 + lr) * 40 + lh * 16];
            fb[nt].q[0] = *(const v4u*)base;
            fb[nt].q[1] = *(const v4u*)(base + 8);
        }
#pragma unroll
        for (int mt = 0; mt < 2; ++mt)
#pragma unroll
            for (int nt = 0; nt < 4; ++nt)
                acc[mt][nt] = __builtin_amdgcn_wmma_f32_16x16x32_bf16(
                    false, fa[mt].v, false, fb[nt].v, (short)0, acc[mt][nt], false, false);

        // fragment reads must complete before next iteration's TDM overwrites buf
        asm volatile("s_wait_dscnt 0x0" ::: "memory");
    }

#pragma unroll
    for (int mt = 0; mt < 2; ++mt)
#pragma unroll
        for (int nt = 0; nt < 4; ++nt)
#pragma unroll
            for (int r = 0; r < 8; ++r) {
                int row = mBase + wm * 32 + mt * 16 + r + 8 * lh;
                int col = nBase + wn * 64 + nt * 16 + lr;
                float v = acc[mt][nt][r];
                if (Cb)        Cb[(size_t)row * ldc + col] = f2b(v);
                else if (beta) Cf[(size_t)row * ldc + col] += v;
                else           Cf[(size_t)row * ldc + col] = v;
            }
}

// =====================================================================
// 4) rotary/xpos: q (scaled, *scale), k (/scale), v transposed copy
// =====================================================================
__global__ void rotary_kernel(const unsigned short* __restrict__ proj,
                              unsigned short* __restrict__ q,
                              unsigned short* __restrict__ k,
                              unsigned short* __restrict__ vt) {
    const int n = blockIdx.x, b = blockIdx.y, d = threadIdx.x;
    const int j = d & 31;
    const float pos  = (float)n;
    const float invf = __powf(10000.0f, -(float)(2 * j) / 64.0f);
    const float f = pos * invf;
    const float c = __cosf(f), s = __sinf(f);
    const float power = (pos - (float)(NSEQ / 2)) / 512.0f;
    const float sb = ((float)(2 * j) + 0.4f * 64.0f) / (1.4f * 64.0f);
    const float sc  = __powf(sb, power);
    const float isc = 1.0f / sc;
    const int   partner = (d < 32) ? d + 32 : d - 32;
    const float sign    = (d < 32) ? -1.0f : 1.0f;
    const unsigned short* prow = proj + ((size_t)b * NSEQ + n) * FUSED;

    {
        float t  = b2f(prow[HEADS * DHEAD + d]);
        float tp = b2f(prow[HEADS * DHEAD + partner]);
        k[((size_t)b * NSEQ + n) * DHEAD + d] = f2b(t * c * isc + sign * tp * s * isc);
    }
    vt[((size_t)b * DHEAD + d) * NSEQ + n] = prow[HEADS * DHEAD + DHEAD + d];
    for (int h = 0; h < HEADS; ++h) {
        float t  = b2f(prow[h * DHEAD + d]) * 0.125f;       // DHEAD^-0.5
        float tp = b2f(prow[h * DHEAD + partner]) * 0.125f;
        q[(((size_t)b * HEADS + h) * NSEQ + n) * DHEAD + d] =
            f2b(t * c * sc + sign * tp * s * sc);
    }
}

// =====================================================================
// 5) flash attention (causal, MQA). grid (32 mblocks, B*H), block 128.
// =====================================================================
__global__ void attn_kernel(const unsigned short* __restrict__ qg,
                            const unsigned short* __restrict__ kg,
                            const unsigned short* __restrict__ vtg,
                            unsigned short* __restrict__ outcat) {
    const int mblk = blockIdx.x;
    const int b = blockIdx.y >> 5;
    const int h = blockIdx.y & 31;
    const int wave = threadIdx.x >> 5;
    const int lane = threadIdx.x & 31;
    const int lr = lane & 15, lh = lane >> 4;
    const int qb = mblk * 64 + wave * 16;

    __shared__ unsigned short pst[4][16 * 40];
    unsigned short* myp = pst[wave];

    const unsigned short* qrow = qg + (((size_t)(b * HEADS + h) * NSEQ) + qb + lr) * DHEAD;
    Frag a0, a1;
    {
        const int kb0 = lh * 16;
        a0.q[0] = *(const v4u*)(qrow + kb0);
        a0.q[1] = *(const v4u*)(qrow + kb0 + 8);
        a1.q[0] = *(const v4u*)(qrow + 32 + kb0);
        a1.q[1] = *(const v4u*)(qrow + 32 + kb0 + 8);
    }

    v8f acc[4];
    float m_i[8], l_i[8];
#pragma unroll
    for (int a = 0; a < 4; ++a) acc[a] = zero8();
#pragma unroll
    for (int r = 0; r < 8; ++r) { m_i[r] = -3.0e38f; l_i[r] = 0.0f; }

    const unsigned short* kbase = kg  + (size_t)b * NSEQ * DHEAD;
    const unsigned short* vbase = vtg + (size_t)b * DHEAD * NSEQ;

    const int jend = qb + 16;
    for (int jt = 0; jt < jend; jt += 32) {
        int j0 = jt + lr;       if (j0 > NSEQ - 1) j0 = NSEQ - 1;
        int j1 = jt + 16 + lr;  if (j1 > NSEQ - 1) j1 = NSEQ - 1;
        const unsigned short* kr0 = kbase + (size_t)j0 * DHEAD;
        const unsigned short* kr1 = kbase + (size_t)j1 * DHEAD;
        if (jt + 32 < jend) {   // WGP-scope prefetch of next tile
            __builtin_prefetch(kr0 + 32 * DHEAD, 0, 3);
            __builtin_prefetch(kr1 + 32 * DHEAD, 0, 3);
        }
        Frag kf00, kf01, kf10, kf11;
        kf00.q[0] = *(const v4u*)(kr0 + lh * 16);      kf00.q[1] = *(const v4u*)(kr0 + lh * 16 + 8);
        kf01.q[0] = *(const v4u*)(kr0 + 32 + lh * 16); kf01.q[1] = *(const v4u*)(kr0 + 32 + lh * 16 + 8);
        kf10.q[0] = *(const v4u*)(kr1 + lh * 16);      kf10.q[1] = *(const v4u*)(kr1 + lh * 16 + 8);
        kf11.q[0] = *(const v4u*)(kr1 + 32 + lh * 16); kf11.q[1] = *(const v4u*)(kr1 + 32 + lh * 16 + 8);

        v8f s0 = zero8(), s1 = zero8();
        s0 = __builtin_amdgcn_wmma_f32_16x16x32_bf16(false, a0.v, false, kf00.v, (short)0, s0, false, false);
        s0 = __builtin_amdgcn_wmma_f32_16x16x32_bf16(false, a1.v, false, kf01.v, (short)0, s0, false, false);
        s1 = __builtin_amdgcn_wmma_f32_16x16x32_bf16(false, a0.v, false, kf10.v, (short)0, s1, false, false);
        s1 = __builtin_amdgcn_wmma_f32_16x16x32_bf16(false, a1.v, false, kf11.v, (short)0, s1, false, false);

        float p0[8], p1[8], corr[8];
#pragma unroll
        for (int r = 0; r < 8; ++r) {
            const int irow = qb + r + 8 * lh;
            float v0 = s0[r]; if (jt + lr > irow)      v0 = -3.0e38f;
            float v1 = s1[r]; if (jt + 16 + lr > irow) v1 = -3.0e38f;
            float mx = rowmax16(fmaxf(v0, v1));
            float mnew = fmaxf(m_i[r], mx);
            float cr = __expf(m_i[r] - mnew);
            float e0 = __expf(v0 - mnew);
            float e1 = __expf(v1 - mnew);
            l_i[r] = l_i[r] * cr + rowsum16(e0 + e1);
            m_i[r] = mnew; corr[r] = cr; p0[r] = e0; p1[r] = e1;
        }
#pragma unroll
        for (int a = 0; a < 4; ++a)
#pragma unroll
            for (int r = 0; r < 8; ++r) acc[a][r] *= corr[r];

        // P: C-layout -> A-layout via per-wave LDS patch
#pragma unroll
        for (int r = 0; r < 8; ++r) {
            const int il = r + 8 * lh;
            myp[il * 40 + lr]      = f2b(p0[r]);
            myp[il * 40 + 16 + lr] = f2b(p1[r]);
        }
        __builtin_amdgcn_fence(__ATOMIC_ACQ_REL, "wavefront");
        Frag pa;
        {
            const unsigned short* base = &myp[lr * 40 + lh * 16];
            pa.q[0] = *(const v4u*)base;
            pa.q[1] = *(const v4u*)(base + 8);
        }
        __builtin_amdgcn_fence(__ATOMIC_ACQ_REL, "wavefront");

        int jv = jt + lh * 16; if (jv > NSEQ - 16) jv = NSEQ - 16;
#pragma unroll
        for (int a = 0; a < 4; ++a) {
            const unsigned short* vr = vbase + (size_t)(a * 16 + lr) * NSEQ + jv;
            if (jt + 32 < jend) __builtin_prefetch(vr + 32, 0, 3);
            Frag vf;
            vf.q[0] = *(const v4u*)vr;
            vf.q[1] = *(const v4u*)(vr + 8);
            acc[a] = __builtin_amdgcn_wmma_f32_16x16x32_bf16(
                false, pa.v, false, vf.v, (short)0, acc[a], false, false);
        }
    }

#pragma unroll
    for (int a = 0; a < 4; ++a)
#pragma unroll
        for (int r = 0; r < 8; ++r) {
            const int irow = qb + r + 8 * lh;
            float o = acc[a][r] / l_i[r];
            outcat[((size_t)b * NSEQ + irow) * (HEADS * DHEAD) + h * DHEAD + a * 16 + lr] = f2b(o);
        }
}

// =====================================================================
// 6) SwiGLU: ffh = silu(gate) * ff_x   (from bf16 proj)
// =====================================================================
__global__ void silu_kernel(const unsigned short* __restrict__ proj,
                            unsigned short* __restrict__ ffh) {
    long long base = (long long)blockIdx.x * 1024 + threadIdx.x;
#pragma unroll
    for (int t = 0; t < 4; ++t) {
        long long idx = base + (long long)t * 256;
        long long m = idx >> 13;
        long long i = idx & (FFI - 1);
        const unsigned short* pr = proj + m * FUSED + (HEADS * DHEAD + 2 * DHEAD);
        float fx = b2f(pr[i]);
        float g  = b2f(pr[FFI + i]);
        float sg = g / (1.0f + __expf(-g));
        ffh[idx] = f2b(sg * fx);
    }
}

// =====================================================================
// launcher
// =====================================================================
extern "C" void kernel_launch(void* const* d_in, const int* in_sizes, int n_in,
                              void* d_out, int out_size, void* d_ws, size_t ws_size,
                              hipStream_t stream) {
    const float* x     = (const float*)d_in[0];
    const float* gamma = (const float*)d_in[1];
    const float* Wf    = (const float*)d_in[2];
    const float* Wao   = (const float*)d_in[3];
    const float* Wff   = (const float*)d_in[4];
    float* out = (float*)d_out;

    char* ws = (char*)d_ws;
    unsigned short* xn    = (unsigned short*)ws; ws += (size_t)MTOT * DIM * 2;
    unsigned short* wf16  = (unsigned short*)ws; ws += (size_t)FUSED * DIM * 2;
    unsigned short* wao16 = (unsigned short*)ws; ws += (size_t)DIM * DIM * 2;
    unsigned short* wff16 = (unsigned short*)ws; ws += (size_t)DIM * FFI * 2;
    unsigned short* proj  = (unsigned short*)ws; ws += (size_t)MTOT * FUSED * 2;
    unsigned short* qbuf  = (unsigned short*)ws; ws += (size_t)BATCH * HEADS * NSEQ * DHEAD * 2;
    unsigned short* kbuf  = (unsigned short*)ws; ws += (size_t)BATCH * NSEQ * DHEAD * 2;
    unsigned short* vtbuf = (unsigned short*)ws; ws += (size_t)BATCH * DHEAD * NSEQ * 2;
    unsigned short* oc    = (unsigned short*)ws; ws += (size_t)MTOT * DIM * 2;
    unsigned short* ffh   = (unsigned short*)ws; ws += (size_t)MTOT * FFI * 2;

    ln_kernel<<<MTOT, 256, 0, stream>>>(x, gamma, xn);
    cvt_kernel<<<(unsigned)(((long long)FUSED * DIM + 1023) / 1024), 256, 0, stream>>>(Wf,  wf16,  (long long)FUSED * DIM);
    cvt_kernel<<<(unsigned)(((long long)DIM * DIM + 1023) / 1024),   256, 0, stream>>>(Wao, wao16, (long long)DIM * DIM);
    cvt_kernel<<<(unsigned)(((long long)DIM * FFI + 1023) / 1024),   256, 0, stream>>>(Wff, wff16, (long long)DIM * FFI);
    gemm_kernel<<<dim3(FUSED / 128, MTOT / 128), 256, 0, stream>>>(xn, wf16, DIM, nullptr, proj, FUSED, 0);
    rotary_kernel<<<dim3(NSEQ, BATCH), 64, 0, stream>>>(proj, qbuf, kbuf, vtbuf);
    attn_kernel<<<dim3(NSEQ / 64, BATCH * HEADS), 128, 0, stream>>>(qbuf, kbuf, vtbuf, oc);
    silu_kernel<<<(unsigned)(((long long)MTOT * FFI) / 1024), 256, 0, stream>>>(proj, ffh);
    gemm_kernel<<<dim3(DIM / 128, MTOT / 128), 256, 0, stream>>>(oc, wao16, DIM, out, nullptr, DIM, 0);
    gemm_kernel<<<dim3(DIM / 128, MTOT / 128), 256, 0, stream>>>(ffh, wff16, FFI, out, nullptr, DIM, 1);
}
